// FeatureDecoder_53154515256145
// MI455X (gfx1250) — matrix operand order repaired and checked by reading, not compiled, not run on hardware
//
#include <hip/hip_runtime.h>

typedef _Float16 v16h __attribute__((ext_vector_type(16)));
typedef _Float16 v8h  __attribute__((ext_vector_type(8)));
typedef float    v8f  __attribute__((ext_vector_type(8)));

#define KDIM 512      // f_in
#define MDIM 256      // f_out
#define NF   65536    // Nf
#define NC   16384    // Nc
#define NB   4        // batch
#define NT   64       // N-tile per workgroup
#define LDSK 528      // padded K stride in halves (1056 B: 32B-aligned, bank-friendly)

// ---------------------------------------------------------------------------
// Kernel 0: split W[512][256] (f32) into hi/lo f16, transposed to [o][k].
// ---------------------------------------------------------------------------
__global__ __launch_bounds__(256) void wsplit_kernel(const float* __restrict__ W,
                                                     _Float16* __restrict__ wt_hi,
                                                     _Float16* __restrict__ wt_lo) {
    int i = blockIdx.x * 256 + threadIdx.x;     // over 512*256
    if (i < KDIM * MDIM) {
        int k = i >> 8;          // row of W
        int o = i & 255;         // col of W
        float v = W[i];
        _Float16 h = (_Float16)v;
        _Float16 l = (_Float16)(v - (float)h);
        wt_hi[o * KDIM + k] = h;
        wt_lo[o * KDIM + k] = l;
    }
}

// ---------------------------------------------------------------------------
// Kernel 1: gather + concat + GEMM (+bias) via v_wmma_f32_16x16x32_f16 x3
// (split-precision fp32 emulation). Each block: full M=256, N-tile of 64.
// ---------------------------------------------------------------------------
__global__ __launch_bounds__(512, 1) void gemm_kernel(const float* __restrict__ enc,
                                                      const float* __restrict__ feat,
                                                      const int*   __restrict__ idx,
                                                      const _Float16* __restrict__ wt_hi,
                                                      const _Float16* __restrict__ wt_lo,
                                                      const float* __restrict__ bias,
                                                      float* __restrict__ out) {
    __shared__ __attribute__((aligned(32))) _Float16 xhi[NT][LDSK];
    __shared__ __attribute__((aligned(32))) _Float16 xlo[NT][LDSK];
    __shared__ int sidx[NT];

    const int bb = blockIdx.x >> 10;            // batch (Nf/NT = 1024 blocks/batch)
    const int n0 = (blockIdx.x & 1023) * NT;    // N-tile start
    const int t  = threadIdx.x;                 // 0..511

    if (t < NT) sidx[t] = idx[bb * NF + n0 + t];
    __syncthreads();

    // -------- fill LDS: x[k][n] -> transposed [n][k], split into hi/lo f16
    {
        const int n  = t & (NT - 1);            // 0..63
        const int k0 = t >> 6;                  // 0..7
        const int idxn = sidx[n];
        const float* encb  = enc  + (bb * MDIM) * NF + n0;   // [k][n] slab
        const float* featb = feat + (bb * MDIM) * NC;        // [c][Nc] slab
        for (int k = k0; k < KDIM; k += 8) {
            float v = (k < MDIM) ? encb[k * NF + n]
                                 : featb[(k - MDIM) * NC + idxn];
            _Float16 h = (_Float16)v;
            _Float16 l = (_Float16)(v - (float)h);
            xhi[n][k] = h;
            xlo[n][k] = l;
        }
    }
    __syncthreads();

    // -------- per-wave 16(M) x 64(N) tile
    const int wave    = t >> 5;                 // 0..15 -> M tile
    const int lane    = t & 31;
    const int o_base  = wave * 16;
    const int m       = lane & 15;
    const int halfsel = lane >> 4;              // 0 or 1
    const int koffA   = halfsel * 16;           // A layout: K offset for hi-lanes
    const int kB      = halfsel * 16;           // B layout: K offset for hi-lanes

    const _Float16* wrow_hi = wt_hi + (o_base + m) * KDIM;
    const _Float16* wrow_lo = wt_lo + (o_base + m) * KDIM;

    v8f acc[4] = {v8f{}, v8f{}, v8f{}, v8f{}};

    for (int k0 = 0; k0 < KDIM; k0 += 32) {
        if (k0 + 32 < KDIM) {  // gfx1250 global_prefetch for next A slice
            __builtin_prefetch(wrow_hi + k0 + 32, 0, 3);
            __builtin_prefetch(wrow_lo + k0 + 32, 0, 3);
        }
        // A fragments (16x32 f16 layout): elems 0-15 -> K=koff..koff+15
        v8h a0h = *(const v8h*)(wrow_hi + k0 + koffA);
        v8h a1h = *(const v8h*)(wrow_hi + k0 + koffA + 8);
        v8h a0l = *(const v8h*)(wrow_lo + k0 + koffA);
        v8h a1l = *(const v8h*)(wrow_lo + k0 + koffA + 8);
        v16h ahi = __builtin_shufflevector(a0h, a1h, 0,1,2,3,4,5,6,7,8,9,10,11,12,13,14,15);
        v16h alo = __builtin_shufflevector(a0l, a1l, 0,1,2,3,4,5,6,7,8,9,10,11,12,13,14,15);

        #pragma unroll
        for (int s = 0; s < 4; ++s) {
            // B fragment (32x16 f16): lane n = lane&15; lanes>=16 hold K=16..31
            const int row = s * 16 + m;
            v16h bhi = *(const v16h*)(&xhi[row][k0 + kB]);
            v16h blo = *(const v16h*)(&xlo[row][k0 + kB]);
            acc[s] = __builtin_amdgcn_wmma_f32_16x16x32_f16(false, ahi, false, bhi,
                                                            (short)0, acc[s], false, false);
            acc[s] = __builtin_amdgcn_wmma_f32_16x16x32_f16(false, ahi, false, blo,
                                                            (short)0, acc[s], false, false);
            acc[s] = __builtin_amdgcn_wmma_f32_16x16x32_f16(false, alo, false, bhi,
                                                            (short)0, acc[s], false, false);
        }
    }

    // -------- bias + store. D layout: VGPR r, lanes0-15: M=r; lanes16-31: M=8+r
    float bv[8];
    #pragma unroll
    for (int r = 0; r < 8; ++r) bv[r] = bias[o_base + halfsel * 8 + r];

    #pragma unroll
    for (int s = 0; s < 4; ++s) {
        const int nn = n0 + s * 16 + m;
        #pragma unroll
        for (int r = 0; r < 8; ++r) {
            const int oo = o_base + halfsel * 8 + r;
            out[((bb * MDIM + oo) << 16) + nn] = acc[s][r] + bv[r];
        }
    }
}

// ---------------------------------------------------------------------------
// Kernel 2: per-channel BN stats (deterministic block reduction, no atomics).
// One block per output channel; emits scale/shift.
// ---------------------------------------------------------------------------
__global__ __launch_bounds__(256) void bn_stats_kernel(const float* __restrict__ y,
                                                       const float* __restrict__ gamma,
                                                       const float* __restrict__ beta,
                                                       float* __restrict__ scale,
                                                       float* __restrict__ shift) {
    const int o = blockIdx.x;                   // 0..255
    const int t = threadIdx.x;
    float s = 0.f, s2 = 0.f;
    for (int b = 0; b < NB; ++b) {
        const float4* p = (const float4*)(y + ((b * MDIM + o) << 16));
        for (int i = t; i < NF / 4; i += 256) {
            float4 v = p[i];
            s  += v.x + v.y + v.z + v.w;
            s2 += v.x * v.x + v.y * v.y + v.z * v.z + v.w * v.w;
        }
    }
    __shared__ float rs[256], rs2[256];
    rs[t] = s; rs2[t] = s2;
    __syncthreads();
    for (int w = 128; w > 0; w >>= 1) {
        if (t < w) { rs[t] += rs[t + w]; rs2[t] += rs2[t + w]; }
        __syncthreads();
    }
    if (t == 0) {
        const float inv_cnt = 1.0f / (float)(NB * NF);
        float mean = rs[0] * inv_cnt;
        float var  = rs2[0] * inv_cnt - mean * mean;
        float sc   = gamma[o] * rsqrtf(var + 1e-5f);
        scale[o] = sc;
        shift[o] = beta[o] - mean * sc;
    }
}

// ---------------------------------------------------------------------------
// Kernel 3: in-place normalize, float4.
// ---------------------------------------------------------------------------
__global__ __launch_bounds__(256) void bn_apply_kernel(float4* __restrict__ y,
                                                       const float* __restrict__ scale,
                                                       const float* __restrict__ shift) {
    const unsigned i = blockIdx.x * 256u + threadIdx.x;   // over 16,777,216 float4
    const int o = (i >> 14) & 255;                        // 16384 float4 per (b,o) row
    const float sc = scale[o], sh = shift[o];
    float4 v = y[i];
    v.x = v.x * sc + sh;
    v.y = v.y * sc + sh;
    v.z = v.z * sc + sh;
    v.w = v.w * sc + sh;
    y[i] = v;
}

// ---------------------------------------------------------------------------
extern "C" void kernel_launch(void* const* d_in, const int* in_sizes, int n_in,
                              void* d_out, int out_size, void* d_ws, size_t ws_size,
                              hipStream_t stream) {
    const float* feature = (const float*)d_in[0];   // [4,256,16384,1]
    const float* enc     = (const float*)d_in[1];   // [4,256,65536,1]
    const int*   idx     = (const int*)  d_in[2];   // [4,65536,1]
    const float* W       = (const float*)d_in[3];   // [512,256]
    const float* bias    = (const float*)d_in[4];   // [256]
    const float* gamma   = (const float*)d_in[5];   // [256]
    const float* beta    = (const float*)d_in[6];   // [256]
    float* out = (float*)d_out;                     // [4,256,65536,1]

    _Float16* wt_hi = (_Float16*)d_ws;              // [256][512] f16
    _Float16* wt_lo = wt_hi + MDIM * KDIM;          // [256][512] f16
    float* scale = (float*)(wt_lo + MDIM * KDIM);   // [256]
    float* shift = scale + MDIM;                    // [256]

    wsplit_kernel<<<(KDIM * MDIM + 255) / 256, 256, 0, stream>>>(W, wt_hi, wt_lo);

    dim3 ggrid(NB * (NF / NT));                     // 4096 blocks
    gemm_kernel<<<ggrid, 512, 0, stream>>>(enc, feature, idx, wt_hi, wt_lo, bias, out);

    bn_stats_kernel<<<MDIM, 256, 0, stream>>>(out, gamma, beta, scale, shift);

    bn_apply_kernel<<<(NB * MDIM * NF / 4) / 256, 256, 0, stream>>>((float4*)out, scale, shift);
}
